// GINClassifier_2276332667278
// MI455X (gfx1250) — matrix operand order repaired and checked by reading, not compiled, not run on hardware
//
#include <hip/hip_runtime.h>
#include <hip/hip_bf16.h>

typedef __attribute__((ext_vector_type(16))) __bf16 v16bf;
typedef __attribute__((ext_vector_type(8)))  float  v8f;
typedef __attribute__((ext_vector_type(4)))  float  v4f;   // native vector for NT loads

#define N_NODES_C  100000
#define N_EDGES_C  1600000
#define N_GRAPHS_C 512
#define HID_C      128
#define EDGE_DIM_C 16
#define BN_EPS_C   1e-5f

// ---------------------------------------------------------------------------
// Pack a row-major f32 [128x128] weight into bf16 WMMA B-fragment order:
// Bp[((kt*8+nt)*32 + lane)*16 + e] = bf16(W[kt*32 + (lane>=16?8:0) + (e&7) + (e>=8?16:0)][nt*16 + lane%16])
// so each lane's 16-element fragment is one contiguous 32B load.
// ---------------------------------------------------------------------------
__global__ __launch_bounds__(256) void pack_w_bf16(const float* __restrict__ W,
                                                   unsigned short* __restrict__ Bp) {
    int t = blockIdx.x * 256 + threadIdx.x;          // 0..16383
    int e    = t & 15;
    int lane = (t >> 4) & 31;
    int nt   = (t >> 9) & 7;
    int kt   = (t >> 12) & 3;
    int n = nt * 16 + (lane & 15);
    int k = kt * 32 + ((lane >> 4) ? 8 : 0) + (e & 7) + ((e >> 3) ? 16 : 0);
    __bf16 v = (__bf16)W[k * HID_C + n];
    Bp[t] = __builtin_bit_cast(unsigned short, v);
}

__global__ __launch_bounds__(256) void copy_f4(const float4* __restrict__ s,
                                               float4* __restrict__ d, int n4) {
    int i = blockIdx.x * 256 + threadIdx.x;
    if (i < n4) d[i] = s[i];
}

__global__ __launch_bounds__(256) void zero_f(float* __restrict__ p, int n) {
    int i = blockIdx.x * 256 + threadIdx.x;
    if (i < n) p[i] = 0.f;
}

// ---------------------------------------------------------------------------
// Fused GINE edge stage: e = attr@le_w + le_b ; msg = relu(x[src]+e);
// atomic scatter-add into agg[dst] (agg pre-initialized to x => "+x" folded).
// 256 threads = 2 edge slots x 128 features; 64 edges per block.
// Barrier-free main loop: the 64B attr row is read directly (L1 broadcasts it
// to the 128 threads of a slot). Streaming operands (edge_attr, edge_index)
// use non-temporal loads so the 192MB L2 stays dedicated to x/agg (the ~16x
// reused gather/scatter working set). Next gather row is prefetched.
// ---------------------------------------------------------------------------
#define EPB 64
__global__ __launch_bounds__(256) void edge_scatter(const float* __restrict__ x,
                                                    float* __restrict__ agg,
                                                    const int* __restrict__ ei,
                                                    const float* __restrict__ eattr,
                                                    const float* __restrict__ lew,
                                                    const float* __restrict__ leb) {
    __shared__ float sW[EDGE_DIM_C * HID_C];     // 8 KB of 320 KB WGP LDS
    __shared__ float sB[HID_C];
    int tid = threadIdx.x;
    for (int i = tid; i < EDGE_DIM_C * HID_C; i += 256) sW[i] = lew[i];
    if (tid < HID_C) sB[tid] = leb[tid];
    __syncthreads();                              // only barrier in the kernel
    int slot = tid >> 7;                          // 0..1
    int f    = tid & 127;
    int ebase = blockIdx.x * EPB;
    #pragma unroll 2
    for (int it = 0; it < EPB / 2; ++it) {
        int e   = ebase + it * 2 + slot;
        int src = __builtin_nontemporal_load(ei + e);
        int dst = __builtin_nontemporal_load(ei + N_EDGES_C + e);
        const v4f* ap = (const v4f*)(eattr + (long long)e * EDGE_DIM_C);
        v4f a0 = __builtin_nontemporal_load(ap + 0);
        v4f a1 = __builtin_nontemporal_load(ap + 1);
        v4f a2 = __builtin_nontemporal_load(ap + 2);
        v4f a3 = __builtin_nontemporal_load(ap + 3);
        if (it + 1 < EPB / 2)                     // prefetch next gather row
            __builtin_prefetch(&x[(long long)ei[e + 2] * HID_C + f], 0, 1);
        float acc = sB[f];
        acc = fmaf(a0.x, sW[ 0 * HID_C + f], acc);
        acc = fmaf(a0.y, sW[ 1 * HID_C + f], acc);
        acc = fmaf(a0.z, sW[ 2 * HID_C + f], acc);
        acc = fmaf(a0.w, sW[ 3 * HID_C + f], acc);
        acc = fmaf(a1.x, sW[ 4 * HID_C + f], acc);
        acc = fmaf(a1.y, sW[ 5 * HID_C + f], acc);
        acc = fmaf(a1.z, sW[ 6 * HID_C + f], acc);
        acc = fmaf(a1.w, sW[ 7 * HID_C + f], acc);
        acc = fmaf(a2.x, sW[ 8 * HID_C + f], acc);
        acc = fmaf(a2.y, sW[ 9 * HID_C + f], acc);
        acc = fmaf(a2.z, sW[10 * HID_C + f], acc);
        acc = fmaf(a2.w, sW[11 * HID_C + f], acc);
        acc = fmaf(a3.x, sW[12 * HID_C + f], acc);
        acc = fmaf(a3.y, sW[13 * HID_C + f], acc);
        acc = fmaf(a3.z, sW[14 * HID_C + f], acc);
        acc = fmaf(a3.w, sW[15 * HID_C + f], acc);
        float msg = x[(long long)src * HID_C + f] + acc;
        msg = msg > 0.f ? msg : 0.f;
        atomicAdd(&agg[(long long)dst * HID_C + f], msg);   // L2-resident target
    }
}

// ---------------------------------------------------------------------------
// WMMA GEMM: out[M,128] = act(Ain)[M,128] @ W[128,128] + bias
//   bn_in   : Ain element a -> relu(a*scale[k] + shift[k])  (BN+ReLU fold)
//   relu_out: relu on store
//   stats   : per-column sum / sumsq of output (pre-activation) for BatchNorm
// One wave computes a 16-row x 128-col strip: 8 v8f accumulators, 32 WMMAs.
// ---------------------------------------------------------------------------
__global__ __launch_bounds__(256) void gemm_wmma(const float* __restrict__ A,
                                                 const unsigned short* __restrict__ Bp,
                                                 const float* __restrict__ bias,
                                                 float* __restrict__ out,
                                                 float* __restrict__ stats,
                                                 const float* __restrict__ scale,
                                                 const float* __restrict__ shift,
                                                 int M, int relu_out, int bn_in) {
    int wave  = threadIdx.x >> 5;
    int lane  = threadIdx.x & 31;
    int strip = blockIdx.x * 8 + wave;
    int m0    = strip * 16;
    if (m0 >= M) return;                           // wave-uniform (EXEC all-1 for WMMA)
    int r    = lane & 15;
    int half = lane >> 4;
    const float* arow = A + (long long)(m0 + r) * HID_C;

    const v8f vzero = {0.f, 0.f, 0.f, 0.f, 0.f, 0.f, 0.f, 0.f};
    v8f acc[8];
    #pragma unroll
    for (int i = 0; i < 8; ++i) acc[i] = vzero;

    #pragma unroll
    for (int kt = 0; kt < 4; ++kt) {
        // A fragment: lane holds row m0+r; K chunks [k0,k0+8) and [k0+16,k0+24)
        // with k0 = kt*32 + (half?8:0)  (CDNA5 16-bit A 16x32 layout)
        int k0 = kt * 32 + (half ? 8 : 0);
        float av[16];
        #pragma unroll
        for (int j = 0; j < 8; ++j) av[j]     = arow[k0 + j];
        #pragma unroll
        for (int j = 0; j < 8; ++j) av[8 + j] = arow[k0 + 16 + j];
        if (bn_in) {
            #pragma unroll
            for (int j = 0; j < 8; ++j) {
                int k = k0 + j;
                float v = fmaf(av[j], scale[k], shift[k]);
                av[j] = v > 0.f ? v : 0.f;
            }
            #pragma unroll
            for (int j = 0; j < 8; ++j) {
                int k = k0 + 16 + j;
                float v = fmaf(av[8 + j], scale[k], shift[k]);
                av[8 + j] = v > 0.f ? v : 0.f;
            }
        }
        v16bf af;
        #pragma unroll
        for (int j = 0; j < 16; ++j) af[j] = (__bf16)av[j];

        #pragma unroll
        for (int nt = 0; nt < 8; ++nt) {
            const v16bf* bp =
                (const v16bf*)(Bp + (((kt * 8 + nt) * 32 + lane) << 4));
            v16bf bf = *bp;                        // contiguous 32B per lane
            acc[nt] = __builtin_amdgcn_wmma_f32_16x16x32_bf16(
                false, af, false, bf, (short)0, acc[nt], false, false);
        }
    }

    // Epilogue: bias, optional ReLU, store, optional BN statistics
    #pragma unroll
    for (int nt = 0; nt < 8; ++nt) {
        int n    = nt * 16 + (lane & 15);
        float bv = bias[n];
        float s = 0.f, ss = 0.f;
        #pragma unroll
        for (int rr = 0; rr < 8; ++rr) {
            float v = acc[nt][rr] + bv;
            if (relu_out) v = v > 0.f ? v : 0.f;
            out[(long long)(m0 + rr + (half ? 8 : 0)) * HID_C + n] = v;
            s += v; ss += v * v;
        }
        if (stats) {
            s  += __shfl_xor(s, 16, 32);           // combine lane halves (wave32)
            ss += __shfl_xor(ss, 16, 32);
            if (half == 0) {
                atomicAdd(&stats[n], s);
                atomicAdd(&stats[HID_C + n], ss);
            }
        }
    }
}

__global__ void bn_finalize(const float* __restrict__ stats,
                            const float* __restrict__ gamma,
                            const float* __restrict__ beta,
                            float* __restrict__ scale, float* __restrict__ shift) {
    int f = threadIdx.x;                           // 128 threads
    float invN = 1.f / (float)N_NODES_C;
    float mean = stats[f] * invN;
    float var  = stats[HID_C + f] * invN - mean * mean;
    float inv  = rsqrtf(var + BN_EPS_C);
    float sc   = gamma[f] * inv;
    scale[f] = sc;
    shift[f] = beta[f] - mean * sc;
}

__global__ __launch_bounds__(256) void pool_kernel(const float* __restrict__ h,
                                                   const int* __restrict__ batch,
                                                   float* __restrict__ pooled) {
    long long gid = (long long)blockIdx.x * 256 + threadIdx.x;
    int n = (int)(gid >> 7), f = (int)(gid & 127);
    if (n < N_NODES_C) {
        float v = __builtin_nontemporal_load(h + (long long)n * HID_C + f);
        atomicAdd(&pooled[batch[n] * HID_C + f], v);    // 256KB target, L2-hot
    }
}

__global__ __launch_bounds__(64) void classifier(const float* __restrict__ pooled,
                                                 const float* __restrict__ Wc1,
                                                 const float* __restrict__ bc1,
                                                 const float* __restrict__ Wc2,
                                                 const float* __restrict__ bc2,
                                                 float* __restrict__ out) {
    __shared__ float sh[64];
    int g = blockIdx.x, j = threadIdx.x;
    float acc = bc1[j];
    #pragma unroll 4
    for (int k = 0; k < HID_C; ++k)
        acc = fmaf(pooled[g * HID_C + k], Wc1[k * 64 + j], acc);
    sh[j] = acc > 0.f ? acc : 0.f;
    __syncthreads();
    if (j < 2) {
        float o = bc2[j];
        #pragma unroll 4
        for (int k = 0; k < 64; ++k) o = fmaf(sh[k], Wc2[k * 2 + j], o);
        out[g * 2 + j] = o;
    }
}

// ---------------------------------------------------------------------------
extern "C" void kernel_launch(void* const* d_in, const int* in_sizes, int n_in,
                              void* d_out, int out_size, void* d_ws, size_t ws_size,
                              hipStream_t stream) {
    const float* x     = (const float*)d_in[0];
    const int*   ei    = (const int*)d_in[1];
    const float* eattr = (const float*)d_in[2];
    const int*   batch = (const int*)d_in[3];
    const float* le1w = (const float*)d_in[4];  const float* le1b = (const float*)d_in[5];
    const float* W11  = (const float*)d_in[6];  const float* b11  = (const float*)d_in[7];
    const float* g1   = (const float*)d_in[8];  const float* bt1  = (const float*)d_in[9];
    const float* W12  = (const float*)d_in[10]; const float* b12  = (const float*)d_in[11];
    const float* le2w = (const float*)d_in[12]; const float* le2b = (const float*)d_in[13];
    const float* W21  = (const float*)d_in[14]; const float* b21  = (const float*)d_in[15];
    const float* g2   = (const float*)d_in[16]; const float* bt2  = (const float*)d_in[17];
    const float* W22  = (const float*)d_in[18]; const float* b22  = (const float*)d_in[19];
    const float* Wc1  = (const float*)d_in[20]; const float* bc1  = (const float*)d_in[21];
    const float* Wc2  = (const float*)d_in[22]; const float* bc2  = (const float*)d_in[23];
    float* out = (float*)d_out;

    // Workspace layout (needs ~103 MB): two N*128 f32 ping-pong buffers + small.
    const long long NF = (long long)N_NODES_C * HID_C;       // 12.8M floats
    float* bufA   = (float*)d_ws;
    float* bufB   = bufA + NF;
    float* stats  = bufB + NF;                                // 256 f
    float* scale  = stats + 2 * HID_C;                        // 128 f
    float* shift  = scale + HID_C;                            // 128 f
    float* pooled = shift + HID_C;                            // 512*128 f
    unsigned short* pk = (unsigned short*)(pooled + (long long)N_GRAPHS_C * HID_C);
    unsigned short* W11p = pk;
    unsigned short* W12p = pk + 16384;
    unsigned short* W21p = pk + 2 * 16384;
    unsigned short* W22p = pk + 3 * 16384;

    // Pack MLP weights into bf16 WMMA fragment order
    pack_w_bf16<<<64, 256, 0, stream>>>(W11, W11p);
    pack_w_bf16<<<64, 256, 0, stream>>>(W12, W12p);
    pack_w_bf16<<<64, 256, 0, stream>>>(W21, W21p);
    pack_w_bf16<<<64, 256, 0, stream>>>(W22, W22p);

    const int n4 = (int)(NF / 4);
    const int cgrid = (n4 + 255) / 256;
    const int ggrid = ((N_NODES_C / 16) + 7) / 8;             // 782 blocks, 8 waves each

    // ---- conv1: agg=A, h1=B, conv1 out=A ----
    copy_f4<<<cgrid, 256, 0, stream>>>((const float4*)x, (float4*)bufA, n4);
    edge_scatter<<<N_EDGES_C / EPB, 256, 0, stream>>>(x, bufA, ei, eattr, le1w, le1b);
    zero_f<<<1, 256, 0, stream>>>(stats, 2 * HID_C);
    gemm_wmma<<<ggrid, 256, 0, stream>>>(bufA, W11p, b11, bufB, stats,
                                         nullptr, nullptr, N_NODES_C, 0, 0);
    bn_finalize<<<1, 128, 0, stream>>>(stats, g1, bt1, scale, shift);
    gemm_wmma<<<ggrid, 256, 0, stream>>>(bufB, W12p, b12, bufA, nullptr,
                                         scale, shift, N_NODES_C, 1, 1);

    // ---- conv2: agg=B, h1=A, conv2 out=B ----
    copy_f4<<<cgrid, 256, 0, stream>>>((const float4*)bufA, (float4*)bufB, n4);
    edge_scatter<<<N_EDGES_C / EPB, 256, 0, stream>>>(bufA, bufB, ei, eattr, le2w, le2b);
    zero_f<<<1, 256, 0, stream>>>(stats, 2 * HID_C);
    gemm_wmma<<<ggrid, 256, 0, stream>>>(bufB, W21p, b21, bufA, stats,
                                         nullptr, nullptr, N_NODES_C, 0, 0);
    bn_finalize<<<1, 128, 0, stream>>>(stats, g2, bt2, scale, shift);
    gemm_wmma<<<ggrid, 256, 0, stream>>>(bufA, W22p, b22, bufB, nullptr,
                                         scale, shift, N_NODES_C, 1, 1);

    // ---- global_add_pool + classifier ----
    zero_f<<<(N_GRAPHS_C * HID_C + 255) / 256, 256, 0, stream>>>(pooled, N_GRAPHS_C * HID_C);
    pool_kernel<<<(int)((NF + 255) / 256), 256, 0, stream>>>(bufB, batch, pooled);
    classifier<<<N_GRAPHS_C, 64, 0, stream>>>(pooled, Wc1, bc1, Wc2, bc2, out);
}
